// DWARFAttention_36017595744563
// MI455X (gfx1250) — matrix operand order repaired and checked by reading, not compiled, not run on hardware
//
#include <hip/hip_runtime.h>
#include <hip/hip_bf16.h>

// ---------------------------------------------------------------------------
// Problem constants (from reference)
// ---------------------------------------------------------------------------
#define BB 4
#define NN 2048
#define DD 1024
#define HH 16
#define HD 64
#define KK 1024            // GEMM reduction dim
#define MM (BB * NN)       // 8192 GEMM rows
#define NSC 11
#define NTAP 4

typedef __attribute__((ext_vector_type(16))) __bf16 v16bf;
typedef __attribute__((ext_vector_type(8)))  __bf16 v8bf;
typedef __attribute__((ext_vector_type(8)))  float  v8f;

union Frag16 { v16bf v; v8bf h[2]; };

__device__ __forceinline__ v8f wmma_bf16(v16bf a, v16bf b, v8f c) {
    // D = A(16x32 bf16) * B(32x16 bf16) + C(16x16 f32)
    return __builtin_amdgcn_wmma_f32_16x16x32_bf16(
        /*neg_a=*/false, a, /*neg_b=*/false, b,
        /*c_mod=*/(short)0, c, /*reuse_a=*/false, /*reuse_b=*/false);
}

// round-to-nearest-even f32 -> bf16, packed pair into u32
__device__ __forceinline__ unsigned int f2bf2(float lo, float hi) {
    unsigned int u0 = __float_as_uint(lo), u1 = __float_as_uint(hi);
    unsigned int r0 = (u0 + 0x7FFFu + ((u0 >> 16) & 1u)) >> 16;
    unsigned int r1 = (u1 + 0x7FFFu + ((u1 >> 16) & 1u)) >> 16;
    return (r0 & 0xFFFFu) | (r1 << 16);
}

// ---------------------------------------------------------------------------
// One-time f32 -> bf16 conversion (bandwidth-bound prep pass)
// ---------------------------------------------------------------------------
__global__ __launch_bounds__(256) void cvt_bf16(
    const float* __restrict__ s, unsigned short* __restrict__ d, int n)
{
    const int i = (blockIdx.x * 256 + threadIdx.x) * 4;
    if (i < n) {
        float4 v = *(const float4*)(s + i);
        unsigned int* p = (unsigned int*)(d + i);
        p[0] = f2bf2(v.x, v.y);
        p[1] = f2bf2(v.z, v.w);
    }
}

// ---------------------------------------------------------------------------
// WMMA GEMM:  C[M, Ncols] = A_bf16[M,K] * W_bf16[Ncols,K]^T  (+ fused epilogue)
//   mode 0: qkv split -> q/k/v head-major [B,H,N,HD] (f32)
//   mode 1: sigmoid(acc + bias) -> gate [M, D] (f32)
//   mode 2: acc + bias -> dst [M, D] (f32)
// Block: 256 threads (8 waves). Tile 64(M) x 128(N) x 32(K).
// Wave (wm = wid&3, wn = wid>>2) owns four 16x16 D tiles at cols wn*64+nt*16;
// the A fragment is loaded once per k-step and reused for all four WMMAs.
// Double-buffered LDS tiles staged with global_load_async_to_lds_b128:
// tile k+1 is in flight (ASYNCcnt) while tile k feeds the WMMA pipe.
// ---------------------------------------------------------------------------
#define LDT 56  // LDS row stride in halves (112B = 7x16B: b128-aligned,
                // lm*28 mod 64 distinct for all 16 fragment lanes)

__global__ __launch_bounds__(256) void gemm_wmma(
    const unsigned short* __restrict__ A, const unsigned short* __restrict__ W,
    const float* __restrict__ bias, int mode,
    float* __restrict__ o0, float* __restrict__ o1, float* __restrict__ o2)
{
    __shared__ __align__(16) unsigned short As[2][64 * LDT];    // 2 x 7.0 KB
    __shared__ __align__(16) unsigned short Bs[2][128 * LDT];   // 2 x 14.0 KB

    const int t    = threadIdx.x;
    const int lane = t & 31;
    const int wid  = t >> 5;
    const int wm   = wid & 3;        // 0..3 -> M sub-tile
    const int wn   = wid >> 2;       // 0..1 -> 64-wide N span
    const int lm   = lane & 15;
    const int hi   = lane >> 4;      // half-wave selector

    const int rowBase = blockIdx.x * 64;    // into M
    const int colBase = blockIdx.y * 128;   // into Ncols

    // staging maps (16B async chunks)
    const int arow = t >> 2, ak = (t & 3) << 3;   // A: 64 rows x 32 halves
    const int brow = t >> 1, bk = (t & 1) << 4;   // B: 128 rows x 32 halves

    const unsigned short* agp = A + (size_t)(rowBase + arow) * KK + ak;
    const unsigned short* bgp = W + (size_t)(colBase + brow) * KK + bk;

    auto issue = [&](int p, int k0) {
        unsigned ldsA = (unsigned)(size_t)&As[p][arow * LDT + ak];
        unsigned long long gaA = (unsigned long long)(const void*)(agp + k0);
        asm volatile("global_load_async_to_lds_b128 %0, %1, off"
                     :: "v"(ldsA), "v"(gaA) : "memory");
        unsigned ldsB = (unsigned)(size_t)&Bs[p][brow * LDT + bk];
        unsigned long long gaB = (unsigned long long)(const void*)(bgp + k0);
        asm volatile("global_load_async_to_lds_b128 %0, %1, off"
                     :: "v"(ldsB), "v"(gaB) : "memory");
        asm volatile("global_load_async_to_lds_b128 %0, %1, off"
                     :: "v"(ldsB + 16u), "v"(gaB + 16ull) : "memory");
    };

    v8f c[4];
    #pragma unroll
    for (int nt = 0; nt < 4; ++nt) c[nt] = v8f{0.f,0.f,0.f,0.f,0.f,0.f,0.f,0.f};

    issue(0, 0);                                   // prologue: tile 0 in flight

    #pragma unroll 2
    for (int i = 0; i < KK / 32; ++i) {
        const int p = i & 1;
        // my tile-i loads are the only outstanding async ops -> drain them
        asm volatile("s_wait_asynccnt 0x0" ::: "memory");
        __syncthreads();   // all waves: tile i resident, tile i-1 reads done
        if (i + 1 < KK / 32) issue(1 - p, (i + 1) * 32);   // overlap with compute

        // ---- A fragment: lane holds M = wm*16+lm; K runs {kb..kb+7, 16+kb..}
        Frag16 a;
        {
            const unsigned short* base = &As[p][(wm * 16 + lm) * LDT + hi * 8];
            a.h[0] = *(const v8bf*)(base);
            a.h[1] = *(const v8bf*)(base + 16);
        }
        // ---- B fragments first (batched ds_loads), then the WMMA chain
        Frag16 bf[4];
        #pragma unroll
        for (int nt = 0; nt < 4; ++nt) {
            const unsigned short* base =
                &Bs[p][(wn * 64 + nt * 16 + lm) * LDT + hi * 8];
            bf[nt].h[0] = *(const v8bf*)(base);
            bf[nt].h[1] = *(const v8bf*)(base + 16);
        }
        #pragma unroll
        for (int nt = 0; nt < 4; ++nt)
            c[nt] = wmma_bf16(a.v, bf[nt].v, c[nt]);
    }

    // ---- epilogue: element r of c -> M = r + hi*8, N = lm (ISA C/D layout)
    #pragma unroll
    for (int nt = 0; nt < 4; ++nt) {
        const int gc = colBase + wn * 64 + nt * 16 + lm;
        const float bv = bias[gc];
        #pragma unroll
        for (int r = 0; r < 8; ++r) {
            const int m   = rowBase + wm * 16 + r + hi * 8;
            const float v = c[nt][r] + bv;
            if (mode == 0) {
                const int which = gc >> 10;          // 0=q 1=k 2=v
                const int d0    = gc & (DD - 1);
                const int hh    = d0 >> 6;
                const int hd    = d0 & (HD - 1);
                const int bb    = m >> 11;
                const int n     = m & (NN - 1);
                float* dst = (which == 0) ? o0 : (which == 1) ? o1 : o2;
                dst[(((size_t)(bb * HH + hh)) * NN + n) * HD + hd] = v;
            } else if (mode == 1) {
                o0[(size_t)m * DD + gc] =
                    __builtin_amdgcn_rcpf(1.f + __expf(-v));
            } else {
                o0[(size_t)m * DD + gc] = v;
            }
        }
    }
}

// ---------------------------------------------------------------------------
// Attention taps: one thread per (b,h,n). Produces y = gathered * gate (bf16).
// ---------------------------------------------------------------------------
__global__ __launch_bounds__(256) void attn_kernel(
    const float* __restrict__ q, const float* __restrict__ k,
    const float* __restrict__ v, const float* __restrict__ gate,
    const float* __restrict__ scale_gain,     // [NSC,HH]
    const float* __restrict__ W_qscale,       // [NSC,HD]
    const float* __restrict__ identity_bypass,// [HH]
    const float* __restrict__ pos_bias,       // [NSC*NTAP,HH]
    unsigned short* __restrict__ y)
{
    __shared__ float sW [NSC * HD];
    __shared__ float sPB[NSC * NTAP * HH];
    __shared__ float sSG[NSC * HH];
    __shared__ float sBP[HH];

    const int t = threadIdx.x;
    for (int i = t; i < NSC * HD;        i += 256) sW[i]  = W_qscale[i];
    for (int i = t; i < NSC * NTAP * HH; i += 256) sPB[i] = pos_bias[i];
    for (int i = t; i < NSC * HH;        i += 256) sSG[i] = scale_gain[i];
    if (t < HH) sBP[t] = identity_bypass[t];
    __syncthreads();

    const int gid = blockIdx.x * 256 + t;
    const int b = gid >> 15;
    const int h = (gid >> 11) & 15;
    const int n = gid & (NN - 1);

    const size_t head = (size_t)(b * HH + h) * NN;
    const float* kb = k + head * HD;
    const float* vb = v + head * HD;

    float qr[HD];
    {
        const float4* p = (const float4*)(q + (head + n) * HD);
        #pragma unroll
        for (int i = 0; i < 16; ++i) {
            float4 qv = p[i];
            qr[4*i] = qv.x; qr[4*i+1] = qv.y; qr[4*i+2] = qv.z; qr[4*i+3] = qv.w;
        }
    }

    // per-scale gains: softmax over 11 logits
    float g[NSC];
    {
        float mx = -1e30f;
        #pragma unroll
        for (int s = 0; s < NSC; ++s) {
            float dot = 0.f;
            #pragma unroll
            for (int d = 0; d < HD; ++d) dot = fmaf(qr[d], sW[s * HD + d], dot);
            g[s] = dot + sSG[s * HH + h];
            mx = fmaxf(mx, g[s]);
        }
        float sum = 0.f;
        #pragma unroll
        for (int s = 0; s < NSC; ++s) { g[s] = __expf(g[s] - mx); sum += g[s]; }
        const float inv = __builtin_amdgcn_rcpf(sum);
        #pragma unroll
        for (int s = 0; s < NSC; ++s) g[s] *= inv;
    }

    // offset-0 dot (shared by every scale's tau=0 tap and the bypass)
    float dot0 = 0.f;
    {
        const float4* p = (const float4*)(kb + (size_t)n * HD);
        #pragma unroll
        for (int i = 0; i < 16; ++i) {
            float4 kv = p[i];
            dot0 = fmaf(qr[4*i], kv.x, fmaf(qr[4*i+1], kv.y,
                   fmaf(qr[4*i+2], kv.z, fmaf(qr[4*i+3], kv.w, dot0))));
        }
    }

    const float D4c[4] = {0.4829629131445341f, 0.8365163037378079f,
                          0.2241438680420134f, -0.1294095225512604f};
    float out[HD];
    #pragma unroll
    for (int d = 0; d < HD; ++d) out[d] = 0.f;
    float z = 0.f;

    #pragma unroll
    for (int j = 0; j < NSC; ++j) {
        const int dlt = 1 << j;
        const float gj = g[j];
        #pragma unroll
        for (int tau = 0; tau < NTAP; ++tau) {
            const int offset = dlt * tau;
            if (offset >= NN) continue;        // compile-time pruned
            const float hc = D4c[tau];
            const float bias_jt = sPB[(j * NTAP + tau) * HH + h];
            const bool has = (n >= offset);
            float dot;
            if (offset == 0) {
                dot = dot0;
            } else if (has) {
                dot = 0.f;
                const float4* p = (const float4*)(kb + (size_t)(n - offset) * HD);
                #pragma unroll
                for (int i = 0; i < 16; ++i) {
                    float4 kv = p[i];
                    dot = fmaf(qr[4*i], kv.x, fmaf(qr[4*i+1], kv.y,
                          fmaf(qr[4*i+2], kv.z, fmaf(qr[4*i+3], kv.w, dot))));
                }
            } else {
                dot = 0.f;                      // k_s is zero-padded
            }
            const float xx = dot + bias_jt;
            const float qk = (xx > 0.f) ? (xx + 1.f) : __expf(xx);  // elu(x)+1
            z = fmaf(gj * fabsf(hc), qk, z);
            if (has) {
                const float co = gj * hc * qk;
                const float4* p = (const float4*)(vb + (size_t)(n - offset) * HD);
                #pragma unroll
                for (int i = 0; i < 16; ++i) {
                    float4 vv = p[i];
                    out[4*i]   = fmaf(co, vv.x, out[4*i]);
                    out[4*i+1] = fmaf(co, vv.y, out[4*i+1]);
                    out[4*i+2] = fmaf(co, vv.z, out[4*i+2]);
                    out[4*i+3] = fmaf(co, vv.w, out[4*i+3]);
                }
            }
        }
    }

    // identity bypass (offset-0)
    {
        const float lf = (dot0 > 0.f) ? (dot0 + 1.f) : __expf(dot0);
        const float bp = log1pf(__expf(sBP[h]));   // softplus
        const float co = bp * lf;
        const float4* p = (const float4*)(vb + (size_t)n * HD);
        #pragma unroll
        for (int i = 0; i < 16; ++i) {
            float4 vv = p[i];
            out[4*i]   = fmaf(co, vv.x, out[4*i]);
            out[4*i+1] = fmaf(co, vv.y, out[4*i+1]);
            out[4*i+2] = fmaf(co, vv.z, out[4*i+2]);
            out[4*i+3] = fmaf(co, vv.w, out[4*i+3]);
        }
        z += co;
    }

    const float inv = __builtin_amdgcn_rcpf(z + 1e-6f);
    const size_t base = ((size_t)(b * NN + n)) * DD + h * HD;
    const float* gp = gate + base;
    unsigned int* yp = (unsigned int*)(y + base);
    #pragma unroll
    for (int i = 0; i < 32; ++i) {
        const float a = out[2*i]   * inv * gp[2*i];
        const float c = out[2*i+1] * inv * gp[2*i+1];
        yp[i] = f2bf2(a, c);
    }
}

// ---------------------------------------------------------------------------
extern "C" void kernel_launch(void* const* d_in, const int* in_sizes, int n_in,
                              void* d_out, int out_size, void* d_ws, size_t ws_size,
                              hipStream_t stream) {
    (void)in_sizes; (void)n_in; (void)out_size; (void)ws_size;
    const float* x         = (const float*)d_in[0];
    const float* W_qkv     = (const float*)d_in[1];
    const float* b_qkv     = (const float*)d_in[2];
    const float* W_out     = (const float*)d_in[3];
    const float* b_out     = (const float*)d_in[4];
    const float* W_gate    = (const float*)d_in[5];
    const float* b_gate    = (const float*)d_in[6];
    const float* sc_gain   = (const float*)d_in[7];
    const float* W_qscale  = (const float*)d_in[8];
    const float* id_bypass = (const float*)d_in[9];
    const float* pos_bias  = (const float*)d_in[10];

    const size_t S = (size_t)BB * NN * DD;   // 8388608 elements per [B,N,D]
    char* w = (char*)d_ws;
    float*          qb    = (float*)w;                 w += S * 4;   // q f32
    float*          kb    = (float*)w;                 w += S * 4;   // k f32
    float*          vb    = (float*)w;                 w += S * 4;   // v f32
    float*          gb    = (float*)w;                 w += S * 4;   // gate f32
    unsigned short* xbf   = (unsigned short*)w;        w += S * 2;   // x bf16
    unsigned short* ybf   = (unsigned short*)w;        w += S * 2;   // y bf16
    unsigned short* wqkvb = (unsigned short*)w;        w += (size_t)3 * DD * KK * 2;
    unsigned short* wgtb  = (unsigned short*)w;        w += (size_t)DD * KK * 2;
    unsigned short* woutb = (unsigned short*)w;

    dim3 blk(256);
    // 0) one-time bf16 conversions (x and all weights)
    cvt_bf16<<<dim3((int)(S / 1024)), blk, 0, stream>>>(x, xbf, (int)S);
    cvt_bf16<<<dim3(3 * DD * KK / 1024), blk, 0, stream>>>(W_qkv, wqkvb, 3 * DD * KK);
    cvt_bf16<<<dim3(DD * KK / 1024), blk, 0, stream>>>(W_gate, wgtb, DD * KK);
    cvt_bf16<<<dim3(DD * KK / 1024), blk, 0, stream>>>(W_out, woutb, DD * KK);
    // 1) qkv projection -> head-major q/k/v
    gemm_wmma<<<dim3(MM / 64, (3 * DD) / 128), blk, 0, stream>>>(
        xbf, wqkvb, b_qkv, 0, qb, kb, vb);
    // 2) gate = sigmoid(x @ W_gate^T + b_gate)
    gemm_wmma<<<dim3(MM / 64, DD / 128), blk, 0, stream>>>(
        xbf, wgtb, b_gate, 1, gb, nullptr, nullptr);
    // 3) wavelet attention taps -> y = gathered * gate (bf16)
    attn_kernel<<<dim3((BB * HH * NN) / 256), blk, 0, stream>>>(
        qb, kb, vb, gb, sc_gain, W_qscale, id_bypass, pos_bias, ybf);
    // 4) result = y @ W_out^T + b_out
    gemm_wmma<<<dim3(MM / 64, DD / 128), blk, 0, stream>>>(
        ybf, woutb, b_out, 2, (float*)d_out, nullptr, nullptr);
}
